// Deformable_Attention_3D_33011118637451
// MI455X (gfx1250) — matrix operand order repaired and checked by reading, not compiled, not run on hardware
//
#include <hip/hip_runtime.h>
#include <hip/hip_bf16.h>

typedef _Float16 f16;
typedef __attribute__((ext_vector_type(16))) _Float16 v16h;
typedef __attribute__((ext_vector_type(8)))  _Float16 v8h;
typedef __attribute__((ext_vector_type(8)))  float    v8f;

// ---- WMMA fragment index helpers (CDNA5 ISA 7.12.2, wave32) -----------------
// A (16x32 f16, MxK): lane L -> row m = L&15 ; element e -> K = (L>>4)*16 + e
// B (32x16 f16, KxN): lane L -> col n = L&15 ; element e -> K = (L>>4)*16 + e
// C/D (16x16 f32):    lane L -> col n = L&15 ; vgpr r  -> row m = (L>>4)*8 + r
__device__ __forceinline__ v8f wmma_f16(const v16h& a, const v16h& b, const v8f& c) {
  return __builtin_amdgcn_wmma_f32_16x16x32_f16(false, a, false, b, (short)0, c,
                                                false, false);
}

// A fragment from a contiguous f16 row (K-major): two aligned 16B chunks.
__device__ __forceinline__ v16h load_a_row(const f16* row, int lane) {
  const int kb = (lane >> 4) << 4;
  const v8h lo = *(const v8h*)(row + kb);
  const v8h hi = *(const v8h*)(row + 8 + kb);
  return __builtin_shufflevector(lo, hi, 0, 1, 2, 3, 4, 5, 6, 7,
                                 8, 9, 10, 11, 12, 13, 14, 15);
}

// B fragment from a channel-contiguous (transposed) tensor: one aligned 32B load.
__device__ __forceinline__ v16h load_b_col(const f16* col, int lane) {
  return *(const v16h*)(col + ((lane >> 4) << 4));
}

// ============================================================================
// Kernel 0: one-shot convert + transpose of reused GEMM operands.
//   qin16t[b][p][c] = (f16)Qf[b][c][p]   (B-operand of Q-proj, K-contiguous)
//   w{q,k,v,o}16    = f16 copies of weights (already K-contiguous rows)
// ============================================================================
__global__ void __launch_bounds__(256)
k_cvt(const float* __restrict__ Qf, const float* __restrict__ wq,
      const float* __restrict__ wk, const float* __restrict__ wv,
      const float* __restrict__ wo, f16* __restrict__ qin16t,
      f16* __restrict__ wq16, f16* __restrict__ wk16,
      f16* __restrict__ wv16, f16* __restrict__ wo16) {
  int i = blockIdx.x * 256 + threadIdx.x;
  if (i < 1048576) {
    // i = (b*4096 + p)*128 + c  -> coalesced writes, strided reads (one-shot)
    const int c  = i & 127;
    const int p  = (i >> 7) & 4095;
    const int b  = i >> 19;
    qin16t[i] = (f16)Qf[((size_t)b * 128 + c) * 4096 + p];
    return;
  }
  i -= 1048576;
  if (i < 16384) { wq16[i] = (f16)wq[i]; return; }
  i -= 16384;
  if (i < 16384) { wk16[i] = (f16)wk[i]; return; }
  i -= 16384;
  if (i < 16384) { wv16[i] = (f16)wv[i]; return; }
  i -= 16384;
  if (i < 16384) wo16[i] = (f16)wo[i];
}

// ============================================================================
// Kernel 1: Q projection.  Q[b,o,p] = sum_c wq[o,c]*Qf[b,c,p] + bq[o]
// Each wave: one 16-row o-tile x TWO adjacent 16-col p-tiles (shared A
// fragment, two independent WMMA chains -> hides WMMA->WMMA RAW hazard).
// Outputs: q32 [b][o][p] (conv branch), q16t [b][p][o] (attention, v8h packed).
// ============================================================================
__global__ void __launch_bounds__(256)
k_qproj(const f16* __restrict__ qin16t, const f16* __restrict__ wq16,
        const float* __restrict__ bq, float* __restrict__ q32,
        f16* __restrict__ q16t) {
  const int lane = threadIdx.x & 31;
  const int wid  = threadIdx.x >> 5;
  int tile = blockIdx.x * 8 + wid;            // 0..2047 : b x ot x ptp(128)
  const int b  = tile >> 10;
  tile &= 1023;
  const int ot  = tile >> 7;                  // 0..7
  const int ptp = tile & 127;                 // pair of p-tiles
  const int lm = lane & 15;
  const int hb = lane >> 4;

  const f16* wrow  = wq16 + (ot * 16 + lm) * 128;
  const f16* qcol0 = qin16t + ((size_t)b * 4096 + ptp * 32 + lm) * 128;
  const f16* qcol1 = qcol0 + 16 * 128;

  v8f acc0 = {}, acc1 = {};
#pragma unroll
  for (int kc = 0; kc < 128; kc += 32) {
    __builtin_prefetch(wrow + kc + 32, 0, 3);
    const v16h a   = load_a_row(wrow + kc, lane);
    const v16h bf0 = load_b_col(qcol0 + kc, lane);
    const v16h bf1 = load_b_col(qcol1 + kc, lane);
    acc0 = wmma_f16(a, bf0, acc0);
    acc1 = wmma_f16(a, bf1, acc1);
  }
  const int p0 = ptp * 32 + lm;
  v8h pack0, pack1;
#pragma unroll
  for (int r = 0; r < 8; ++r) {
    const int o = ot * 16 + (hb << 3) + r;
    const float bias = bq[o];
    const float v0 = acc0[r] + bias;
    const float v1 = acc1[r] + bias;
    q32[((size_t)b * 128 + o) * 4096 + p0]      = v0;
    q32[((size_t)b * 128 + o) * 4096 + p0 + 16] = v1;
    pack0[r] = (f16)v0;
    pack1[r] = (f16)v1;
  }
  f16* qt = q16t + ((size_t)b * 4096 + p0) * 128 + ot * 16 + (hb << 3);
  *(v8h*)(qt)            = pack0;   // 16B store, p0
  *(v8h*)(qt + 16 * 128) = pack1;   // 16B store, p0+16
}

// ============================================================================
// Kernel 2: depthwise 3x3x3 conv, stride 2, pad 1 on Q reshaped (8,32,16^3)
// -> off1 (8,32,8,8,8).  One thread per output element (131072).
// ============================================================================
__global__ void __launch_bounds__(256)
k_conv(const float* __restrict__ q32, const float* __restrict__ wdw,
       const float* __restrict__ bdw, float* __restrict__ off1) {
  const int idx = blockIdx.x * blockDim.x + threadIdx.x;
  if (idx >= 8 * 32 * 512) return;
  const int s  = idx & 511;
  const int c  = (idx >> 9) & 31;
  const int gb = idx >> 14;
  const int b = gb >> 2, g = gb & 3;
  const int ow = s & 7, oh = (s >> 3) & 7, od = s >> 6;

  const float* in = q32 + ((size_t)(b * 128 + g * 32 + c)) * 4096;
  const float* w  = wdw + c * 27;
  float acc = bdw[c];
#pragma unroll
  for (int kz = 0; kz < 3; ++kz) {
    const int z = od * 2 - 1 + kz;
    if (z < 0 || z >= 16) continue;
#pragma unroll
    for (int ky = 0; ky < 3; ++ky) {
      const int y = oh * 2 - 1 + ky;
      if (y < 0 || y >= 16) continue;
#pragma unroll
      for (int kx = 0; kx < 3; ++kx) {
        const int x = ow * 2 - 1 + kx;
        if (x < 0 || x >= 16) continue;
        acc += w[kz * 9 + ky * 3 + kx] * in[z * 256 + y * 16 + x];
      }
    }
  }
  off1[idx] = acc;
}

// ============================================================================
// Kernel 3: LayerNorm(32ch) + exact GELU + proj->3 + tanh*ORF/8 + ref grid
// + trilinear zero-pad sample of KV (16^3) -> xs16t[b][n][c] (channel-contig).
// One thread per (group-batch, sample) = 4096 threads; 4x v8h packed stores.
// ============================================================================
__global__ void __launch_bounds__(128)
k_offsample(const float* __restrict__ off1, const float* __restrict__ lnw,
            const float* __restrict__ lnb, const float* __restrict__ wproj,
            const float* __restrict__ KVf, f16* __restrict__ xs16t) {
  const int idx = blockIdx.x * blockDim.x + threadIdx.x;
  if (idx >= 8 * 512) return;
  const int s  = idx & 511;
  const int gb = idx >> 9;
  const int b = gb >> 2, g = gb & 3;

  const float* col = off1 + (size_t)gb * 16384 + s;
  float v[32];
  float sum = 0.f, sum2 = 0.f;
#pragma unroll
  for (int c = 0; c < 32; ++c) {
    v[c] = col[c * 512];
    sum += v[c];
    sum2 += v[c] * v[c];
  }
  const float mu   = sum * (1.f / 32.f);
  const float var  = sum2 * (1.f / 32.f) - mu * mu;
  const float rinv = rsqrtf(var + 1e-5f);

  float p0 = 0.f, p1 = 0.f, p2 = 0.f;
#pragma unroll
  for (int c = 0; c < 32; ++c) {
    const float x  = (v[c] - mu) * rinv * lnw[c] + lnb[c];
    const float gx = 0.5f * x * (1.f + erff(x * 0.70710678118654752f)); // exact gelu
    p0 += wproj[c]      * gx;
    p1 += wproj[32 + c] * gx;
    p2 += wproj[64 + c] * gx;
  }
  // tanh * ORF(2.0) * offset_range(1/8) = tanh * 0.25  (channels: 0->z,1->y,2->x)
  const float offz = tanhf(p0) * 0.25f;
  const float offy = tanhf(p1) * 0.25f;
  const float offx = tanhf(p2) * 0.25f;

  const int ow = s & 7, oh = (s >> 3) & 7, od = s >> 6;
  const float gz = offz + ((od + 0.5f) * 0.25f - 1.f);
  const float gy = offy + ((oh + 0.5f) * 0.25f - 1.f);
  const float gx = offx + ((ow + 0.5f) * 0.25f - 1.f);

  // unnormalize: (g+1)*0.5*(dim-1), dim = 16
  const float iz = (gz + 1.f) * 7.5f;
  const float iy = (gy + 1.f) * 7.5f;
  const float ix = (gx + 1.f) * 7.5f;

  const float z0f = floorf(iz), y0f = floorf(iy), x0f = floorf(ix);
  const float tz = iz - z0f, ty = iy - y0f, tx = ix - x0f;
  const int z0 = (int)z0f, y0 = (int)y0f, x0 = (int)x0f;

  int   coff[8];
  float cw[8];
  int k = 0;
#pragma unroll
  for (int dz = 0; dz < 2; ++dz)
#pragma unroll
    for (int dy = 0; dy < 2; ++dy)
#pragma unroll
      for (int dx = 0; dx < 2; ++dx) {
        const int zi = z0 + dz, yi = y0 + dy, xi = x0 + dx;
        const bool ok = (zi >= 0) & (zi < 16) & (yi >= 0) & (yi < 16) &
                        (xi >= 0) & (xi < 16);
        const float w = (dz ? tz : 1.f - tz) * (dy ? ty : 1.f - ty) *
                        (dx ? tx : 1.f - tx) * (ok ? 1.f : 0.f);
        const int zc = min(max(zi, 0), 15), yc = min(max(yi, 0), 15),
                  xc = min(max(xi, 0), 15);
        coff[k] = zc * 256 + yc * 16 + xc;
        cw[k]   = w;
        ++k;
      }

  const float* kvbase = KVf + ((size_t)b * 128 + g * 32) * 4096;
  f16* dst = xs16t + ((size_t)b * 512 + s) * 128 + g * 32;
#pragma unroll
  for (int cb = 0; cb < 32; cb += 8) {
    v8h pack;
#pragma unroll
    for (int j = 0; j < 8; ++j) {
      const float* img = kvbase + (cb + j) * 4096;
      float acc = 0.f;
#pragma unroll
      for (int q = 0; q < 8; ++q) acc += cw[q] * img[coff[q]];
      pack[j] = (f16)acc;
    }
    *(v8h*)(dst + cb) = pack;           // 16B contiguous store
  }
}

// ============================================================================
// Kernel 4: fused K and V projections on sampled features.
//   kht[b][n][c] (transposed, packed v8h stores)  -- QK^T B operand
//   vh [b][c][n] (row-major)                      -- PV^T B operand
// Two independent WMMA chains (K and V) already hide the RAW hazard.
// ============================================================================
__global__ void __launch_bounds__(256)
k_kvproj(const f16* __restrict__ xs16t, const f16* __restrict__ wk16,
         const float* __restrict__ bk, const f16* __restrict__ wv16,
         const float* __restrict__ bv, f16* __restrict__ kht,
         f16* __restrict__ vh) {
  const int lane = threadIdx.x & 31;
  const int wid  = threadIdx.x >> 5;
  int tile = blockIdx.x * 8 + wid;           // 0..511
  const int b  = tile >> 8;
  const int ot = (tile >> 5) & 7;
  const int pt = tile & 31;
  const int lm = lane & 15;
  const int hb = lane >> 4;

  const f16* wkrow = wk16 + (ot * 16 + lm) * 128;
  const f16* wvrow = wv16 + (ot * 16 + lm) * 128;
  const f16* xcol  = xs16t + ((size_t)b * 512 + pt * 16 + lm) * 128;

  v8f acck = {}, accv = {};
#pragma unroll
  for (int kc = 0; kc < 128; kc += 32) {
    const v16h ak = load_a_row(wkrow + kc, lane);
    const v16h av = load_a_row(wvrow + kc, lane);
    const v16h bf = load_b_col(xcol + kc, lane);
    acck = wmma_f16(ak, bf, acck);
    accv = wmma_f16(av, bf, accv);
  }
  const int n = pt * 16 + lm;
  v8h packk;
#pragma unroll
  for (int r = 0; r < 8; ++r) {
    const int o = ot * 16 + (hb << 3) + r;
    packk[r] = (f16)(acck[r] + bk[o]);
    vh[((size_t)b * 128 + o) * 512 + n] = (f16)(accv[r] + bv[o]);
  }
  *(v8h*)(kht + ((size_t)b * 512 + n) * 128 + ot * 16 + (hb << 3)) = packk;
}

// ============================================================================
// Kernel 5: attention per (batch*head, 16-row query tile).
// A(Q^T) = one v16h load (lanes 0-15, K pad 16->32), B(K) = one v16h load (lanes 0-15),
// scores -> f32 LDS -> full-wave softmax -> probs f16 -> P @ V^T via WMMA
// with even/odd dual accumulators (hazard hiding).
// ============================================================================
__global__ void __launch_bounds__(32)
k_attn(const f16* __restrict__ q16t, const f16* __restrict__ kht,
       const f16* __restrict__ vh, f16* __restrict__ att16t) {
  __shared__ float sS[16 * 512];
  __shared__ f16   sP[16 * 512];

  const int lane = threadIdx.x;
  const int tile = blockIdx.x;        // 16 bh x 256 m-tiles
  const int bh = tile >> 8;
  const int mt = tile & 255;
  const int b = bh >> 3, h = bh & 7;
  const int lm = lane & 15;
  const int hb = lane >> 4;

  // A = Qh^T tile: rows = 16 query positions, K = 16 head channels (pad to 32)
  v16h aq;
  {
    const f16* qrow = q16t + ((size_t)b * 4096 + mt * 16 + lm) * 128 + h * 16;
#pragma unroll
    for (int e = 0; e < 16; ++e) aq[e] = (f16)0.f;
    if (hb == 0)
      aq = *(const v16h*)qrow;
  }

  // scores: 32 n-tiles of 16 -> LDS  (B: K=channel contiguous in kht;
  // each iteration's WMMA chain is independent -> no hazard stall)
  const f16* kbase = kht + ((size_t)b * 512) * 128 + h * 16;
  for (int nt = 0; nt < 32; ++nt) {
    v16h bkf;
#pragma unroll
    for (int e = 0; e < 16; ++e) bkf[e] = (f16)0.f;
    if (hb == 0)                      // lanes 16-31 carry the K=16..31 zero pad
      bkf = *(const v16h*)(kbase + (size_t)(nt * 16 + lm) * 128);
    v8f sc = {};
    sc = wmma_f16(aq, bkf, sc);
#pragma unroll
    for (int r = 0; r < 8; ++r)
      sS[((hb << 3) + r) * 512 + nt * 16 + lm] = sc[r];
  }
  __syncthreads();

  // softmax over 512 samples per row, scale = HC^-0.5 = 0.25.
  // Lane L handles half a row: row = L&15, cols [hb*256, hb*256+256).
  {
    float* rowp = sS + lm * 512 + hb * 256;
    float mx = -3.4e38f;
    for (int i = 0; i < 256; ++i) mx = fmaxf(mx, rowp[i] * 0.25f);
    mx = fmaxf(mx, __shfl_xor(mx, 16, 32));
    float sum = 0.f;
    for (int i = 0; i < 256; ++i) {
      const float e = __expf(rowp[i] * 0.25f - mx);
      rowp[i] = e;
      sum += e;
    }
    sum += __shfl_xor(sum, 16, 32);
    const float inv = 1.f / sum;
    f16* prow = sP + lm * 512 + hb * 256;
    for (int i = 0; i < 256; ++i) prow[i] = (f16)(rowp[i] * inv);
  }
  __syncthreads();

  // out[m][c] = sum_n P[m][n] * V[c][n]  (A = P tile from LDS, B = V^T tile)
  // even/odd K-step accumulators break the WMMA->WMMA RAW chain.
  const size_t kvb = ((size_t)b * 128 + h * 16) * 512;
  v8f acc0 = {}, acc1 = {};
  for (int kt = 0; kt < 16; kt += 2) {
    const v16h ap0  = load_a_row(sP + lm * 512 + kt * 32, lane);
    const v16h ap1  = load_a_row(sP + lm * 512 + (kt + 1) * 32, lane);
    const v16h bvf0 = *(const v16h*)(vh + kvb + (size_t)lm * 512 +
                                     kt * 32 + (hb << 4));
    const v16h bvf1 = *(const v16h*)(vh + kvb + (size_t)lm * 512 +
                                     (kt + 1) * 32 + (hb << 4));
    acc0 = wmma_f16(ap0, bvf0, acc0);
    acc1 = wmma_f16(ap1, bvf1, acc1);
  }
  // store transposed: att16t[b][p][c]  (c = lm, p = mt*16 + (hb*8+r))
#pragma unroll
  for (int r = 0; r < 8; ++r) {
    const int mm = (hb << 3) + r;
    att16t[((size_t)b * 4096 + mt * 16 + mm) * 128 + h * 16 + lm] =
        (f16)(acc0[r] + acc1[r]);
  }
}

// ============================================================================
// Kernel 6: output projection y[b,o,p] = sum_c wo[o,c]*out[b,c,p] + bo[o]
// Two p-tiles per wave (shared A fragment, dual accumulators).
// ============================================================================
__global__ void __launch_bounds__(256)
k_oproj(const f16* __restrict__ att16t, const f16* __restrict__ wo16,
        const float* __restrict__ bo, float* __restrict__ y) {
  const int lane = threadIdx.x & 31;
  const int wid  = threadIdx.x >> 5;
  int tile = blockIdx.x * 8 + wid;           // 0..2047
  const int b  = tile >> 10;
  tile &= 1023;
  const int ot  = tile >> 7;
  const int ptp = tile & 127;
  const int lm = lane & 15;
  const int hb = lane >> 4;

  const f16* wrow  = wo16 + (ot * 16 + lm) * 128;
  const f16* acol0 = att16t + ((size_t)b * 4096 + ptp * 32 + lm) * 128;
  const f16* acol1 = acol0 + 16 * 128;

  v8f acc0 = {}, acc1 = {};
#pragma unroll
  for (int kc = 0; kc < 128; kc += 32) {
    __builtin_prefetch(wrow + kc + 32, 0, 3);
    const v16h a   = load_a_row(wrow + kc, lane);
    const v16h bf0 = load_b_col(acol0 + kc, lane);
    const v16h bf1 = load_b_col(acol1 + kc, lane);
    acc0 = wmma_f16(a, bf0, acc0);
    acc1 = wmma_f16(a, bf1, acc1);
  }
  const int p0 = ptp * 32 + lm;
#pragma unroll
  for (int r = 0; r < 8; ++r) {
    const int o = ot * 16 + (hb << 3) + r;
    const float bias = bo[o];
    y[((size_t)b * 128 + o) * 4096 + p0]      = acc0[r] + bias;
    y[((size_t)b * 128 + o) * 4096 + p0 + 16] = acc1[r] + bias;
  }
}

// ============================================================================
extern "C" void kernel_launch(void* const* d_in, const int* in_sizes, int n_in,
                              void* d_out, int out_size, void* d_ws, size_t ws_size,
                              hipStream_t stream) {
  (void)in_sizes; (void)n_in; (void)out_size; (void)ws_size;

  const float* Qf    = (const float*)d_in[0];   // (2,128,16,16,16)
  const float* KVf   = (const float*)d_in[1];   // (2,128,16,16,16)
  const float* wq    = (const float*)d_in[2];
  const float* bq    = (const float*)d_in[3];
  const float* wdw   = (const float*)d_in[4];
  const float* bdw   = (const float*)d_in[5];
  const float* lnw   = (const float*)d_in[6];
  const float* lnb   = (const float*)d_in[7];
  const float* wproj = (const float*)d_in[8];
  const float* wk    = (const float*)d_in[9];
  const float* bk    = (const float*)d_in[10];
  const float* wv    = (const float*)d_in[11];
  const float* bv    = (const float*)d_in[12];
  const float* wo    = (const float*)d_in[13];
  const float* bo    = (const float*)d_in[14];
  float* y = (float*)d_out;

  char* ws = (char*)d_ws;
  float* q32    = (float*)(ws);                      // 4 MB   (2,128,4096) f32
  f16*   q16t   = (f16*)  (ws + 4194304);            // 2 MB   Q f16, [b][p][c]
  f16*   qin16t = (f16*)  (ws + 6291456);            // 2 MB   Q_feature f16, [b][p][c]
  f16*   att16t = (f16*)  (ws + 8388608);            // 2 MB   attn out f16, [b][p][c]
  f16*   xs16t  = (f16*)  (ws + 10485760);           // 256 KB xs f16, [b][n][c]
  f16*   kht    = (f16*)  (ws + 10747904);           // 256 KB K f16, [b][n][c]
  f16*   vh     = (f16*)  (ws + 11010048);           // 256 KB V f16, [b][c][n]
  float* off1   = (float*)(ws + 11272192);           // 512 KB conv out f32
  f16*   wq16   = (f16*)  (ws + 11796480);           // 32 KB
  f16*   wk16   = (f16*)  (ws + 11829248);           // 32 KB
  f16*   wv16   = (f16*)  (ws + 11862016);           // 32 KB
  f16*   wo16   = (f16*)  (ws + 11894784);           // 32 KB

  // 0) one-shot convert + transpose of reused operands
  k_cvt<<<4352, 256, 0, stream>>>(Qf, wq, wk, wv, wo, qin16t, wq16, wk16, wv16, wo16);
  // 1) Q projection (WMMA): 2048 wave-tiles (2 p-tiles each)
  k_qproj<<<256, 256, 0, stream>>>(qin16t, wq16, bq, q32, q16t);
  // 2) depthwise conv stride-2
  k_conv<<<512, 256, 0, stream>>>(q32, wdw, bdw, off1);
  // 3) LN + GELU + proj + tanh + trilinear sample
  k_offsample<<<32, 128, 0, stream>>>(off1, lnw, lnb, wproj, KVf, xs16t);
  // 4) fused K/V projection (WMMA): 512 wave-tiles
  k_kvproj<<<64, 256, 0, stream>>>(xs16t, wk16, bk, wv16, bv, kht, vh);
  // 5) attention (WMMA QK^T + softmax + WMMA PV): 4096 wave-tiles
  k_attn<<<4096, 32, 0, stream>>>(q16t, kht, vh, att16t);
  // 6) output projection (WMMA, 2 p-tiles per wave)
  k_oproj<<<256, 256, 0, stream>>>(att16t, wo16, bo, y);
}
